// SimplifiedEvoformer_54219667145444
// MI455X (gfx1250) — compile-verified, matrix operand order checked by reading, not run on hardware
//
#include <hip/hip_runtime.h>
#include <hip/hip_bf16.h>
#include <math.h>

// ---------------------------------------------------------------------------
// SimplifiedEvoformer for MI455X (gfx1250): all GEMMs on v_wmma_f32_16x16x32_bf16
// ---------------------------------------------------------------------------
typedef __bf16 bf16;
typedef __attribute__((ext_vector_type(16))) __bf16 v16bf;
typedef __attribute__((ext_vector_type(8)))  float  v8f;

#define BB 1
#define NN 128
#define LL 256
#define HH 256
#define PP 128
#define NHEADS 8

// K index of vgpr-pair v for a lane half (ISA 7.12.2, 16-bit A 16x32)
__device__ __forceinline__ int frag_kk(int v, int half) {
  return ((v >> 2) << 4) + (half << 3) + ((v & 3) << 1);
}

// ---------------------------------------------------------------------------
// f32 -> bf16 conversion (weights)
// ---------------------------------------------------------------------------
__global__ void f32_to_bf16_kernel(const float* __restrict__ in,
                                   bf16* __restrict__ out, int n) {
  int i = blockIdx.x * blockDim.x + threadIdx.x;
  if (i < n) out[i] = (bf16)in[i];
}

// ---------------------------------------------------------------------------
// Embedding gather: x[row, c] = emb[msa[row], c]; writes f32 + bf16 copies
// ---------------------------------------------------------------------------
__global__ void embed_kernel(const int* __restrict__ msa, const float* __restrict__ emb,
                             float* __restrict__ xf, bf16* __restrict__ xb) {
  int row = blockIdx.x;              // 0 .. N*L-1
  int c   = threadIdx.x;             // 0 .. H-1
  int tok = msa[row];
  float v = emb[tok * HH + c];
  size_t idx = (size_t)row * HH + c;
  xf[idx] = v;
  xb[idx] = (bf16)v;
}

// ---------------------------------------------------------------------------
// WMMA GEMM, templated epilogue:  out = act(A[M,K] @ W[K,N] + bias) (+ residual)
// A,W bf16 row-major; out f32 and/or bf16. M%64==0, N%64==0, K%32==0.
// 256 threads (8 waves), 64x64 tile; wave w -> rows [(w&1)*32,+32), cols [(w>>1)*16,+16).
// W tile staged TRANSPOSED so B-fragment reads are contiguous b128 loads.
// ---------------------------------------------------------------------------
template <bool HB, bool HR, bool OF, bool OB, bool RELU>
__global__ __launch_bounds__(256)
void gemm_bf16_kernel(const bf16* __restrict__ A, const bf16* __restrict__ W,
                      const float* __restrict__ bias, const float* __restrict__ resid,
                      float* __restrict__ outF, bf16* __restrict__ outB,
                      int M, int N, int K) {
  __shared__ __align__(16) bf16 As [64 * 32];   // [m][k]
  __shared__ __align__(16) bf16 WsT[64 * 32];   // [n][k]  (transposed)

  const int tid  = threadIdx.x;
  const int lane = tid & 31;
  const int wave = tid >> 5;
  const int wm   = wave & 1;
  const int wn   = wave >> 1;
  const int half = lane >> 4;
  const int l16  = lane & 15;
  const int m0   = blockIdx.y * 64;
  const int n0   = blockIdx.x * 64;

  v8f acc0 = {};
  v8f acc1 = {};

  for (int k0 = 0; k0 < K; k0 += 32) {
    // Stage A tile 64x32: one 16B chunk per thread, row-major
    {
      int r = tid >> 2, ch = tid & 3;
      *(uint4*)&As[r * 32 + ch * 8] =
          *(const uint4*)&A[(size_t)(m0 + r) * K + k0 + ch * 8];
    }
    // Stage W tile 32x64 transposed -> WsT[n][k]
    {
      int r2 = tid >> 3, ch2 = tid & 7;    // k row r2, n cols ch2*8..+7
      uint4 raw = *(const uint4*)&W[(size_t)(k0 + r2) * N + n0 + ch2 * 8];
      const bf16* e = (const bf16*)&raw;
#pragma unroll
      for (int j = 0; j < 8; ++j)
        WsT[(ch2 * 8 + j) * 32 + r2] = e[j];
    }
    __syncthreads();

    v16bf a0, a1, b;
#pragma unroll
    for (int v = 0; v < 8; ++v) {
      int kk = frag_kk(v, half);
      a0[2 * v]     = As[(wm * 32 + l16) * 32 + kk];
      a0[2 * v + 1] = As[(wm * 32 + l16) * 32 + kk + 1];
      a1[2 * v]     = As[(wm * 32 + 16 + l16) * 32 + kk];
      a1[2 * v + 1] = As[(wm * 32 + 16 + l16) * 32 + kk + 1];
      b[2 * v]      = WsT[(wn * 16 + l16) * 32 + kk];
      b[2 * v + 1]  = WsT[(wn * 16 + l16) * 32 + kk + 1];
    }
    acc0 = __builtin_amdgcn_wmma_f32_16x16x32_bf16(false, a0, false, b,
                                                   (short)0, acc0, false, false);
    acc1 = __builtin_amdgcn_wmma_f32_16x16x32_bf16(false, a1, false, b,
                                                   (short)0, acc1, false, false);
    __syncthreads();
  }

  // Epilogue (specialized; C/D layout row = r + 8*half, col = l16)
  const int colg = n0 + wn * 16 + l16;
  const float bv = HB ? bias[colg] : 0.f;
#pragma unroll
  for (int r = 0; r < 8; ++r) {
    int rowl = r + half * 8;
    {
      int rowg = m0 + wm * 32 + rowl;
      size_t idx = (size_t)rowg * N + colg;
      float v = acc0[r] + bv;
      if (RELU) v = v > 0.f ? v : 0.f;
      if (HR)   v += resid[idx];
      if (OF)   outF[idx] = v;
      if (OB)   outB[idx] = (bf16)v;
    }
    {
      int rowg = m0 + wm * 32 + 16 + rowl;
      size_t idx = (size_t)rowg * N + colg;
      float v = acc1[r] + bv;
      if (RELU) v = v > 0.f ? v : 0.f;
      if (HR)   v += resid[idx];
      if (OF)   outF[idx] = v;
      if (OB)   outB[idx] = (bf16)v;
    }
  }
}

// ---------------------------------------------------------------------------
// Fused multi-head attention, one workgroup per (batch, head).
// Seq len 256, head dim d in {16, 32}, K-dim zero-padded to 32.
// qkv rows: [q(E) | k(E) | v(E)], row stride 3E. Output o: row stride E.
// All staging done in 16-byte chunks (pad boundary is 8-element aligned).
// ---------------------------------------------------------------------------
__global__ __launch_bounds__(256)
void attn_kernel(const bf16* __restrict__ qkv, bf16* __restrict__ o,
                 int h, int d, int E, float scale) {
  __shared__ __align__(16) bf16 Ksh [256 * 32];  // [key][c]   16 KB
  __shared__ __align__(16) bf16 VshT[32 * 256];  // [c][key]   16 KB
  __shared__ __align__(16) bf16 Qsh [16 * 32];   // [q][c]      1 KB
  __shared__ float Ssh[16 * 256];                // scores     16 KB
  __shared__ __align__(16) bf16  Psh[16 * 256];  // probs       8 KB
  __shared__ float Osh[16 * 32];                 //             2 KB
  __shared__ float red[16 * 17];
  __shared__ float rowm[16], rinv[16];

  const int tid   = threadIdx.x;
  const int lane  = tid & 31;
  const int wave  = tid >> 5;
  const int half  = lane >> 4;
  const int l16   = lane & 15;
  const int batch = blockIdx.x / h;
  const int head  = blockIdx.x % h;
  const int E3    = 3 * E;
  const size_t base = (size_t)batch * 256 * E3;
  const uint4 zero4 = make_uint4(0u, 0u, 0u, 0u);

  // Stage K (row-major) and V (transposed) in 16B chunks, zero-padded to 32 ch
  for (int ch = tid; ch < 256 * 4; ch += 256) {
    int l = ch >> 2, c0 = (ch & 3) * 8;
    uint4 kraw = zero4, vraw = zero4;
    if (c0 < d) {
      kraw = *(const uint4*)&qkv[base + (size_t)l * E3 + E     + head * d + c0];
      vraw = *(const uint4*)&qkv[base + (size_t)l * E3 + 2 * E + head * d + c0];
    }
    *(uint4*)&Ksh[l * 32 + c0] = kraw;
    const bf16* e = (const bf16*)&vraw;
#pragma unroll
    for (int j = 0; j < 8; ++j)
      VshT[(c0 + j) * 256 + l] = e[j];
  }
  __syncthreads();

  for (int qt = 0; qt < 16; ++qt) {
    // Stage Q tile 16x32 in 16B chunks (64 chunks; first 64 threads)
    if (tid < 64) {
      int l = tid >> 2, c0 = (tid & 3) * 8;
      uint4 qraw = zero4;
      if (c0 < d)
        qraw = *(const uint4*)&qkv[base + (size_t)(qt * 16 + l) * E3 + head * d + c0];
      *(uint4*)&Qsh[l * 32 + c0] = qraw;
    }
    __syncthreads();

    // Q fragment (identical in every wave)
    v16bf qa;
#pragma unroll
    for (int v = 0; v < 8; ++v) {
      int kk = frag_kk(v, half);
      qa[2 * v]     = Qsh[l16 * 32 + kk];
      qa[2 * v + 1] = Qsh[l16 * 32 + kk + 1];
    }

    // S = scale * Q K^T : wave handles key tiles 2*wave, 2*wave+1
#pragma unroll
    for (int t = 0; t < 2; ++t) {
      int kt = wave * 2 + t;
      v16bf kb;
#pragma unroll
      for (int v = 0; v < 8; ++v) {
        int kk = frag_kk(v, half);
        kb[2 * v]     = Ksh[(kt * 16 + l16) * 32 + kk];
        kb[2 * v + 1] = Ksh[(kt * 16 + l16) * 32 + kk + 1];
      }
      v8f s = {};
      s = __builtin_amdgcn_wmma_f32_16x16x32_bf16(false, qa, false, kb,
                                                  (short)0, s, false, false);
#pragma unroll
      for (int r = 0; r < 8; ++r)
        Ssh[(r + half * 8) * 256 + kt * 16 + l16] = s[r] * scale;
    }
    __syncthreads();

    // Row softmax over 256 keys: 16 threads per row
    {
      int row = tid >> 4, sub = tid & 15;
      float m = -1e30f;
      for (int j = 0; j < 16; ++j)
        m = fmaxf(m, Ssh[row * 256 + sub + j * 16]);
      red[row * 17 + sub] = m;
      __syncthreads();
      if (sub == 0) {
        float mm = red[row * 17];
        for (int j = 1; j < 16; ++j) mm = fmaxf(mm, red[row * 17 + j]);
        rowm[row] = mm;
      }
      __syncthreads();
      float mm = rowm[row];
      float ssum = 0.f;
      for (int j = 0; j < 16; ++j) {
        int c = sub + j * 16;
        float e = __expf(Ssh[row * 256 + c] - mm);
        Ssh[row * 256 + c] = e;
        ssum += e;
      }
      red[row * 17 + sub] = ssum;
      __syncthreads();
      if (sub == 0) {
        float t = 0.f;
        for (int j = 0; j < 16; ++j) t += red[row * 17 + j];
        rinv[row] = 1.0f / t;
      }
      __syncthreads();
      float inv = rinv[row];
      for (int j = 0; j < 16; ++j) {
        int c = sub + j * 16;
        Psh[row * 256 + c] = (bf16)(Ssh[row * 256 + c] * inv);
      }
    }

    // Zero O accumulator tile (16 x 32)
    Osh[tid]       = 0.f;
    Osh[tid + 256] = 0.f;
    __syncthreads();

    // O = P @ V : wave handles key chunk kc = wave (32 keys)
    {
      int kc = wave;
      v16bf pa;
#pragma unroll
      for (int v = 0; v < 8; ++v) {
        int kk = frag_kk(v, half);
        pa[2 * v]     = Psh[l16 * 256 + kc * 32 + kk];
        pa[2 * v + 1] = Psh[l16 * 256 + kc * 32 + kk + 1];
      }
      for (int nt = 0; nt < (d >> 4); ++nt) {
        v16bf vb;
#pragma unroll
        for (int v = 0; v < 8; ++v) {
          int kk = frag_kk(v, half);
          vb[2 * v]     = VshT[(nt * 16 + l16) * 256 + kc * 32 + kk];
          vb[2 * v + 1] = VshT[(nt * 16 + l16) * 256 + kc * 32 + kk + 1];
        }
        v8f oacc = {};
        oacc = __builtin_amdgcn_wmma_f32_16x16x32_bf16(false, pa, false, vb,
                                                       (short)0, oacc, false, false);
#pragma unroll
        for (int r = 0; r < 8; ++r)
          atomicAdd(&Osh[(r + half * 8) * 32 + nt * 16 + l16], oacc[r]);
      }
    }
    __syncthreads();

    // Write O tile (bf16) in 4-element chunks: 16*d elems, d % 4 == 0
    for (int ch = tid; ch < 4 * d; ch += 256) {   // 16 rows * (d/4) chunks
      int row = ch / (d >> 2), c0 = (ch % (d >> 2)) * 4;
      bf16 pk[4];
#pragma unroll
      for (int j = 0; j < 4; ++j) pk[j] = (bf16)Osh[row * 32 + c0 + j];
      *(uint2*)&o[((size_t)batch * 256 + qt * 16 + row) * E + head * d + c0] =
          *(uint2*)pk;
    }
    __syncthreads();
  }
}

// ---------------------------------------------------------------------------
// LayerNorm: one block per row, blockDim == W, dynamic shared = W floats
// ---------------------------------------------------------------------------
template <bool OB>
__global__ void ln_kernel(const float* __restrict__ in, const float* __restrict__ g,
                          const float* __restrict__ b, float* __restrict__ outF,
                          bf16* __restrict__ outB, int W) {
  extern __shared__ float red[];
  int row = blockIdx.x, tid = threadIdx.x;
  float x = in[(size_t)row * W + tid];
  red[tid] = x;
  __syncthreads();
  for (int s = W >> 1; s > 0; s >>= 1) {
    if (tid < s) red[tid] += red[tid + s];
    __syncthreads();
  }
  float mu = red[0] / W;
  __syncthreads();
  float dx = x - mu;
  red[tid] = dx * dx;
  __syncthreads();
  for (int s = W >> 1; s > 0; s >>= 1) {
    if (tid < s) red[tid] += red[tid + s];
    __syncthreads();
  }
  float var = red[0] / W;
  float y = dx * rsqrtf(var + 1e-5f) * g[tid] + b[tid];
  size_t idx = (size_t)row * W + tid;
  outF[idx] = y;
  if (OB) outB[idx] = (bf16)y;
}

// ---------------------------------------------------------------------------
// Pair build: y[i,j,:] = qw1[i,:] + qw2[j,:] + pair_b
// ---------------------------------------------------------------------------
__global__ void pair_build_kernel(const float* __restrict__ qw1,
                                  const float* __restrict__ qw2,
                                  const float* __restrict__ pb,
                                  float* __restrict__ yf, bf16* __restrict__ yb) {
  int row = blockIdx.x;            // i*256 + j
  int p   = threadIdx.x;           // 0..127
  int i = row >> 8, j = row & 255;
  float v = qw1[i * PP + p] + qw2[j * PP + p] + pb[p];
  size_t idx = (size_t)row * PP + p;
  yf[idx] = v;
  yb[idx] = (bf16)v;
}

// ---------------------------------------------------------------------------
// Host-side orchestration
// ---------------------------------------------------------------------------
extern "C" void kernel_launch(void* const* d_in, const int* in_sizes, int n_in,
                              void* d_out, int out_size, void* d_ws, size_t ws_size,
                              hipStream_t stream) {
  const int*   msa        = (const int*)  d_in[0];
  const float* emb        = (const float*)d_in[1];
  const float* pair_w     = (const float*)d_in[2];
  const float* pair_b     = (const float*)d_in[3];
  const float* msa_in_w   = (const float*)d_in[4];
  const float* msa_in_b   = (const float*)d_in[5];
  const float* msa_out_w  = (const float*)d_in[6];
  const float* msa_out_b  = (const float*)d_in[7];
  const float* pr_in_w    = (const float*)d_in[8];
  const float* pr_in_b    = (const float*)d_in[9];
  const float* pr_out_w   = (const float*)d_in[10];
  const float* pr_out_b   = (const float*)d_in[11];
  const float* msa_ffn_w1 = (const float*)d_in[12];
  const float* msa_ffn_b1 = (const float*)d_in[13];
  const float* msa_ffn_w2 = (const float*)d_in[14];
  const float* msa_ffn_b2 = (const float*)d_in[15];
  const float* pr_ffn_w1  = (const float*)d_in[16];
  const float* pr_ffn_b1  = (const float*)d_in[17];
  const float* pr_ffn_w2  = (const float*)d_in[18];
  const float* pr_ffn_b2  = (const float*)d_in[19];
  const float* mn1_g = (const float*)d_in[20];
  const float* mn1_b = (const float*)d_in[21];
  const float* mn2_g = (const float*)d_in[22];
  const float* mn2_b = (const float*)d_in[23];
  const float* pn1_g = (const float*)d_in[24];
  const float* pn1_b = (const float*)d_in[25];
  const float* pn2_g = (const float*)d_in[26];
  const float* pn2_b = (const float*)d_in[27];

  const int M1 = NN * LL;   // 32768 rows, MSA track
  const int M2 = LL * LL;   // 65536 rows, pair track
  const size_t ACT = (size_t)M1 * HH;  // 8,388,608 (== M2 * PP)

  char* ws = (char*)d_ws;
  size_t off = 0;
  auto take = [&](size_t bytes) -> void* {
    void* p = ws + off;
    off += (bytes + 255) & ~(size_t)255;
    return p;
  };
  float* bufA_f32 = (float*)take(ACT * 4);
  float* bufB_f32 = (float*)take(ACT * 4);
  bf16*  bufA_bf  = (bf16*) take(ACT * 2);
  bf16*  qkv_bf   = (bf16*) take((size_t)M1 * 3 * HH * 2);
  bf16*  o_bf     = (bf16*) take(ACT * 2);
  bf16*  hid_bf   = (bf16*) take((size_t)M1 * 4 * HH * 2);
  float* qw1      = (float*)take((size_t)LL * PP * 4);
  float* qw2      = (float*)take((size_t)LL * PP * 4);
  bf16* w_msa_in  = (bf16*)take((size_t)HH * 3 * HH * 2);
  bf16* w_msa_out = (bf16*)take((size_t)HH * HH * 2);
  bf16* w_msa_f1  = (bf16*)take((size_t)HH * 4 * HH * 2);
  bf16* w_msa_f2  = (bf16*)take((size_t)4 * HH * HH * 2);
  bf16* w_pair    = (bf16*)take((size_t)2 * HH * PP * 2);
  bf16* w_pr_in   = (bf16*)take((size_t)PP * 3 * PP * 2);
  bf16* w_pr_out  = (bf16*)take((size_t)PP * PP * 2);
  bf16* w_pr_f1   = (bf16*)take((size_t)PP * 4 * PP * 2);
  bf16* w_pr_f2   = (bf16*)take((size_t)4 * PP * PP * 2);

  auto cvt = [&](const float* s, bf16* d, int n) {
    f32_to_bf16_kernel<<<(n + 255) / 256, 256, 0, stream>>>(s, d, n);
  };
  auto gemm_proj = [&](const bf16* A, const bf16* W, const float* bias,
                       const float* resid, float* oF, int M, int N, int K) {
    dim3 g(N / 64, M / 64);
    gemm_bf16_kernel<true, true, true, false, false>
        <<<g, 256, 0, stream>>>(A, W, bias, resid, oF, nullptr, M, N, K);
  };
  auto gemm_bf = [&](const bf16* A, const bf16* W, const float* bias, bf16* oB,
                     int M, int N, int K) {
    dim3 g(N / 64, M / 64);
    gemm_bf16_kernel<true, false, false, true, false>
        <<<g, 256, 0, stream>>>(A, W, bias, nullptr, nullptr, oB, M, N, K);
  };
  auto gemm_relu = [&](const bf16* A, const bf16* W, const float* bias, bf16* oB,
                       int M, int N, int K) {
    dim3 g(N / 64, M / 64);
    gemm_bf16_kernel<true, false, false, true, true>
        <<<g, 256, 0, stream>>>(A, W, bias, nullptr, nullptr, oB, M, N, K);
  };
  auto gemm_f32 = [&](const bf16* A, const bf16* W, float* oF,
                      int M, int N, int K) {
    dim3 g(N / 64, M / 64);
    gemm_bf16_kernel<false, false, true, false, false>
        <<<g, 256, 0, stream>>>(A, W, nullptr, nullptr, oF, nullptr, M, N, K);
  };
  auto ln = [&](const float* in, const float* g, const float* b,
                float* oF, bf16* oB, int rows, int W) {
    if (oB) ln_kernel<true><<<rows, W, W * sizeof(float), stream>>>(in, g, b, oF, oB, W);
    else    ln_kernel<false><<<rows, W, W * sizeof(float), stream>>>(in, g, b, oF, nullptr, W);
  };

  float* out_msa  = (float*)d_out;          // (1,128,256,256)
  float* out_pair = out_msa + ACT;          // (1,256,256,128)

  // ---- weight conversions ----
  cvt(msa_in_w,   w_msa_in,  HH * 3 * HH);
  cvt(msa_out_w,  w_msa_out, HH * HH);
  cvt(msa_ffn_w1, w_msa_f1,  HH * 4 * HH);
  cvt(msa_ffn_w2, w_msa_f2,  4 * HH * HH);
  cvt(pair_w,     w_pair,    2 * HH * PP);
  cvt(pr_in_w,    w_pr_in,   PP * 3 * PP);
  cvt(pr_out_w,   w_pr_out,  PP * PP);
  cvt(pr_ffn_w1,  w_pr_f1,   PP * 4 * PP);
  cvt(pr_ffn_w2,  w_pr_f2,   4 * PP * PP);

  // ================= MSA track =================
  embed_kernel<<<M1, HH, 0, stream>>>(msa, emb, bufA_f32, bufA_bf);
  gemm_bf(bufA_bf, w_msa_in, msa_in_b, qkv_bf, M1, 3 * HH, HH);
  attn_kernel<<<NN * NHEADS, 256, 0, stream>>>(qkv_bf, o_bf, NHEADS, 32, HH,
                                               0.17677669529663687f);
  gemm_proj(o_bf, w_msa_out, msa_out_b, bufA_f32, bufB_f32, M1, HH, HH);
  ln(bufB_f32, mn1_g, mn1_b, bufA_f32, bufA_bf, M1, HH);
  gemm_relu(bufA_bf, w_msa_f1, msa_ffn_b1, hid_bf, M1, 4 * HH, HH);
  gemm_proj(hid_bf, w_msa_f2, msa_ffn_b2, bufA_f32, bufB_f32, M1, HH, 4 * HH);
  ln(bufB_f32, mn2_g, mn2_b, out_msa, bufA_bf, M1, HH);

  // ================= Pair track =================
  gemm_f32(bufA_bf, w_pair,           qw1, LL, PP, HH);
  gemm_f32(bufA_bf, w_pair + HH * PP, qw2, LL, PP, HH);
  pair_build_kernel<<<M2, PP, 0, stream>>>(qw1, qw2, pair_b, bufA_f32, bufA_bf);
  gemm_bf(bufA_bf, w_pr_in, pr_in_b, qkv_bf, M2, 3 * PP, PP);
  attn_kernel<<<LL * NHEADS, 256, 0, stream>>>(qkv_bf, o_bf, NHEADS, 16, PP, 0.25f);
  gemm_proj(o_bf, w_pr_out, pr_out_b, bufA_f32, bufB_f32, M2, PP, PP);
  ln(bufB_f32, pn1_g, pn1_b, bufA_f32, bufA_bf, M2, PP);
  gemm_relu(bufA_bf, w_pr_f1, pr_ffn_b1, hid_bf, M2, 4 * PP, PP);
  gemm_proj(hid_bf, w_pr_f2, pr_ffn_b2, bufA_f32, bufB_f32, M2, PP, 4 * PP);
  ln(bufB_f32, pn2_g, pn2_b, out_pair, nullptr, M2, PP);

  (void)in_sizes; (void)n_in; (void)out_size; (void)ws_size;
}
